// VectorQuantizer_31542239821999
// MI455X (gfx1250) — matrix operand order repaired and checked by reading, not compiled, not run on hardware
//
#include <hip/hip_runtime.h>

typedef __attribute__((ext_vector_type(16))) _Float16 v16h;
typedef __attribute__((ext_vector_type(8)))  float    v8f;

#define NUM_EMB   1024
#define EMB_DIM   64
#define HW        4096            // 64*64
#define NFLAT     65536           // 16*64*64
#define OUT_ELEMS 4194304         // 16*64*64*64
#define EMB_HALFS (NUM_EMB * EMB_DIM)   // 65536 halves = 128 KB

// ---------------------------------------------------------------------------
// Kernel 1: codebook prep — f32 -> f16 copy, pre-scaled row norms
// (-0.5*||e||^2, so it can seed the WMMA C operand), zero the accumulator.
// ---------------------------------------------------------------------------
__global__ void vq_prep(const float* __restrict__ emb,
                        float* __restrict__ nhnorm,
                        _Float16* __restrict__ embh,
                        float* __restrict__ acc) {
    const int k = blockIdx.x * blockDim.x + threadIdx.x;
    if (k == 0) *acc = 0.0f;
    if (k < NUM_EMB) {
        const float* e = emb + (size_t)k * EMB_DIM;
        _Float16* eh = embh + (size_t)k * EMB_DIM;
        float s = 0.0f;
#pragma unroll
        for (int c = 0; c < EMB_DIM; ++c) {
            float v = e[c];
            s += v * v;
            eh[c] = (_Float16)v;
        }
        nhnorm[k] = -0.5f * s;
    }
}

// ---------------------------------------------------------------------------
// Kernel 2: WMMA distance GEMM + row-wise argmin.
// Block = 256 threads = 8 wave32s; each wave owns a 16-row tile of flat x.
// Codebook (f16) + scaled norms staged once per block into LDS via CDNA5
// global_load_async_to_lds_b128 (ASYNCcnt), WMMA B operands via ds_load_b128.
//
// Score(n,k) = x_n.e_k - 0.5*||e_k||^2  computed entirely inside WMMA by
// seeding C with -0.5*||e||^2; argmin distance == argmax score.
// ---------------------------------------------------------------------------
__global__ void __launch_bounds__(256)
vq_argmin(const float* __restrict__ x,       // [16,64,64,64] = [b][c][hw]
          const _Float16* __restrict__ embh, // [1024][64] row-major
          const float* __restrict__ nhnorm,  // [1024]  (-0.5*||e||^2)
          int* __restrict__ idx) {           // [65536]
    __shared__ _Float16 sEmb[EMB_HALFS];     // 128 KB
    __shared__ float    sNorm[NUM_EMB];      //   4 KB

    const int tid = threadIdx.x;

    // ---- async stage: 132 KB codebook+norms -> LDS (no VGPR round trip) ----
    {
        // low 32 bits of the generic address == LDS byte offset
        uint32_t ldsEmb  = (uint32_t)(uintptr_t)(&sEmb[0]);
        uint32_t ldsNorm = (uint32_t)(uintptr_t)(&sNorm[0]);
        const char* gEmb  = (const char*)embh;
        const char* gNorm = (const char*)nhnorm;
#pragma unroll
        for (int i = 0; i < 32; ++i) {                 // 32 * 4 KB = 128 KB
            uint32_t off = (uint32_t)(tid * 16 + i * 4096);
            asm volatile("global_load_async_to_lds_b128 %0, %1, off"
                         :: "v"(ldsEmb + off), "v"(gEmb + off) : "memory");
        }
        {
            uint32_t off = (uint32_t)(tid * 16);       // 256 * 16 B = 4 KB
            asm volatile("global_load_async_to_lds_b128 %0, %1, off"
                         :: "v"(ldsNorm + off), "v"(gNorm + off) : "memory");
        }
        asm volatile("s_wait_asynccnt 0x0" ::: "memory");
    }
    __syncthreads();

    const int lane  = tid & 31;
    const int wave  = tid >> 5;
    const int nBase = blockIdx.x * 128 + wave * 16;
    const int row   = lane & 15;
    const int half  = lane >> 4;

    // x[n][c] lives at inputs[b*C*HW + c*HW + hw], n = b*HW + hw
    const int n  = nBase + row;
    const int b  = n >> 12;
    const int hw = n & (HW - 1);
    const float* xp = x + (size_t)b * (EMB_DIM * HW) + hw;

    // A operand: contiguous k order, matching B (not the ISA table order).
    // slot i (VGPR i/2, half i&1):  K = i + 16*halfsel
    v16h a0, a1;
#pragma unroll
    for (int i = 0; i < 16; ++i) {
        const int v = i >> 1;
        const int c = i + (half << 4); (void)v;
        a0[i] = (_Float16)xp[(size_t)c * HW];
        a1[i] = (_Float16)xp[(size_t)(c + 32) * HW];
    }

    float best[8];
    int   bidx[8];
#pragma unroll
    for (int r = 0; r < 8; ++r) { best[r] = -3.0e38f; bidx[r] = 0; }

    const int j = lane & 15;      // output column N = code-within-chunk
#pragma unroll 2
    for (int kc = 0; kc < NUM_EMB / 16; ++kc) {
        const int codeBase = kc * 16;
        const int code     = codeBase + j;
        // B operand: lanes 0-15 hold K=0..15, lanes 16-31 hold K=16..31,
        // K linear across the 16 half-slots -> contiguous 32B slices of a row.
        const _Float16* ep = &sEmb[code * EMB_DIM];
        v16h b0 = *(const v16h*)(ep + 16 * half);        // C = 0..31 chunk
        v16h b1 = *(const v16h*)(ep + 32 + 16 * half);   // C = 32..63 chunk

        // Seed C with -0.5*||e_code||^2 (same per-lane column for all 8 rows)
        const float nk = sNorm[code];
        v8f acc = { nk, nk, nk, nk, nk, nk, nk, nk };
        acc = __builtin_amdgcn_wmma_f32_16x16x32_f16(
                  false, a0, false, b0, (short)0, acc, false, false);
        acc = __builtin_amdgcn_wmma_f32_16x16x32_f16(
                  false, a1, false, b1, (short)0, acc, false, false);

#pragma unroll
        for (int r = 0; r < 8; ++r) {
            if (acc[r] > best[r]) { best[r] = acc[r]; bidx[r] = code; } // keep 1st
        }
    }

    // Butterfly max-reduce across the 16 lanes of each C/D half-group.
    // VGPR r: lanes 0-15 -> row r, lanes 16-31 -> row r+8.
#pragma unroll
    for (int off = 8; off >= 1; off >>= 1) {
#pragma unroll
        for (int r = 0; r < 8; ++r) {
            const float ov = __shfl_xor(best[r], off, 32);
            const int   oi = __shfl_xor(bidx[r], off, 32);
            if (ov > best[r] || (ov == best[r] && oi < bidx[r])) {
                best[r] = ov; bidx[r] = oi;
            }
        }
    }
#pragma unroll
    for (int r = 0; r < 8; ++r) {
        if ((lane & 15) == r) idx[nBase + r + (half << 3)] = bidx[r];
    }
}

// ---------------------------------------------------------------------------
// Kernel 3: coalesced gather-write of quantized output (in [B,C,H,W] layout)
// + squared-error reduction for the loss.
// ---------------------------------------------------------------------------
__global__ void __launch_bounds__(256)
vq_output(const float* __restrict__ x,
          const float* __restrict__ emb,
          const int* __restrict__ idx,
          float* __restrict__ out,
          float* __restrict__ acc) {
    __shared__ float red[256];
    const size_t o  = (size_t)blockIdx.x * 256 + threadIdx.x;  // < OUT_ELEMS
    const int    b  = (int)(o >> 18);
    const int    c  = (int)(o >> 12) & 63;
    const int    hw = (int)o & (HW - 1);
    const int    n  = (b << 12) + hw;

    const float q = emb[(size_t)idx[n] * EMB_DIM + c];
    out[o] = q;                       // quantized_st == quantized numerically
    const float d = q - x[o];
    red[threadIdx.x] = d * d;
    __syncthreads();
#pragma unroll
    for (int s = 128; s > 0; s >>= 1) {
        if (threadIdx.x < s) red[threadIdx.x] += red[threadIdx.x + s];
        __syncthreads();
    }
    if (threadIdx.x == 0) atomicAdd(acc, red[0]);
}

// ---------------------------------------------------------------------------
// Kernel 4: loss = (1 + 0.25) * mean((q-x)^2)
// ---------------------------------------------------------------------------
__global__ void vq_loss(const float* __restrict__ acc, float* __restrict__ loss) {
    *loss = 1.25f * (*acc) / (float)OUT_ELEMS;
}

// ---------------------------------------------------------------------------
extern "C" void kernel_launch(void* const* d_in, const int* in_sizes, int n_in,
                              void* d_out, int out_size, void* d_ws, size_t ws_size,
                              hipStream_t stream) {
    const float* inputs = (const float*)d_in[0];   // [16,64,64,64]
    const float* emb    = (const float*)d_in[1];   // [1024,64]
    float* out = (float*)d_out;                    // [4194304 quantized] + [1 loss]

    // workspace layout
    char* ws = (char*)d_ws;
    float*    nhn  = (float*)ws;                              //   4 KB
    _Float16* embh = (_Float16*)(ws + 4096);                  // 128 KB
    int*      idx  = (int*)(ws + 4096 + 131072);              // 256 KB
    float*    acc  = (float*)(ws + 4096 + 131072 + 262144);   //   4 B

    vq_prep<<<4, 256, 0, stream>>>(emb, nhn, embh, acc);
    vq_argmin<<<NFLAT / 128, 256, 0, stream>>>(inputs, embh, nhn, idx);
    vq_output<<<OUT_ELEMS / 256, 256, 0, stream>>>(inputs, emb, idx, out, acc);
    vq_loss<<<1, 1, 0, stream>>>(acc, out + OUT_ELEMS);
}
